// RNN_88261577933327
// MI455X (gfx1250) — matrix operand order repaired and checked by reading, not compiled, not run on hardware
//
#include <hip/hip_runtime.h>
#include <hip/hip_bf16.h>

// ---------------------------------------------------------------------------
// Elman RNN for MI455X (gfx1250, wave32, WMMA)
//   T=2048 steps, L=512 in/out, H=4096 hidden
//   Phase 1: xh = x @ W_i2h^T + b_i2h          (bf16 WMMA GEMM)
//   Phase 2: h_t = tanh(xh_t + W_h2h h_{t-1} + b_h2h)  (persistent kernel,
//            GEMV-as-WMMA with h broadcast across M, grid barrier per step,
//            W_h2h kept bf16 (32 MB) so it stays resident in the 192 MB L2)
//   Phase 3: out = h_all @ W_h2o^T + b_h2o     (bf16 WMMA GEMM)
// ---------------------------------------------------------------------------

typedef __bf16 v16bf __attribute__((ext_vector_type(16)));
typedef __bf16 v8bf  __attribute__((ext_vector_type(8)));
typedef float  v8f   __attribute__((ext_vector_type(8)));

static constexpr int   TT = 2048;
static constexpr int   LL = 512;
static constexpr int   HH = 4096;
static constexpr int   NB = 64;    // persistent blocks for recurrence

// ---- workspace layout (bytes, all 256-aligned) ----
static constexpr size_t OFF_XH   = 0;                                    // [T][H] f32
static constexpr size_t SZ_XH    = (size_t)TT * HH * 4;
static constexpr size_t OFF_HALL = OFF_XH + SZ_XH;                       // [T+1][H] bf16 (row0 = h_{-1} = 0)
static constexpr size_t SZ_HALL  = (size_t)(TT + 1) * HH * 2;
static constexpr size_t OFF_W2   = OFF_HALL + SZ_HALL;                   // W_h2h bf16 [H][H]
static constexpr size_t SZ_W2    = (size_t)HH * HH * 2;
static constexpr size_t OFF_XBF  = OFF_W2 + SZ_W2;                       // x bf16 [T][L]
static constexpr size_t SZ_XBF   = (size_t)TT * LL * 2;
static constexpr size_t OFF_W1   = OFF_XBF + SZ_XBF;                     // W_i2h bf16 [H][L]
static constexpr size_t SZ_W1    = (size_t)HH * LL * 2;
static constexpr size_t OFF_W3   = OFF_W1 + SZ_W1;                       // W_h2o bf16 [L][H]
static constexpr size_t SZ_W3    = (size_t)LL * HH * 2;
static constexpr size_t OFF_BAR  = OFF_W3 + SZ_W3;                       // grid barrier counter

__global__ __launch_bounds__(256)
void cvt_f32_bf16(const float* __restrict__ src, __bf16* __restrict__ dst, int n) {
    int i = blockIdx.x * 256 + threadIdx.x;
    if (i < n) dst[i] = (__bf16)src[i];
}

// D = A[M,K] * B^T (B stored row-major as W[n][k]) + bias, one 16x16 tile per wave.
// A operand: lane<16 -> row m0+lane, k0..k0+15
//            lane>=16 -> row m0+lane-16, k0+16..k0+31
// B operand: lane<16 -> W[n0+lane][k0..k0+15]; lane>=16 -> W[n0+lane-16][k0+16..k0+31]
__global__ __launch_bounds__(256)
void wmma_gemm_bias(const __bf16* __restrict__ A, int lda,
                    const __bf16* __restrict__ B, int ldb,
                    const float* __restrict__ bias,
                    float* __restrict__ C, int ldc,
                    int mtiles, int K) {
    const int wave = threadIdx.x >> 5;
    const int lane = threadIdx.x & 31;
    const int tid  = blockIdx.x * 8 + wave;
    const int m0   = (tid % mtiles) * 16;
    const int n0   = (tid / mtiles) * 16;
    const bool hi  = lane >= 16;
    const size_t arow = (size_t)(m0 + (lane & 15)) * lda;
    const size_t brow = (size_t)(n0 + (lane & 15)) * ldb;
    const int aoff = hi ? 16 : 0;
    const int boff = hi ? 16 : 0;
    v8f c = {};
    #pragma unroll 4
    for (int k0 = 0; k0 < K; k0 += 32) {
        v8bf alo = *(const v8bf*)(A + arow + k0 + aoff);
        v8bf ahi = *(const v8bf*)(A + arow + k0 + aoff + 8);
        v16bf a = __builtin_shufflevector(alo, ahi, 0,1,2,3,4,5,6,7,8,9,10,11,12,13,14,15);
        v16bf b = *(const v16bf*)(B + brow + k0 + boff);
        c = __builtin_amdgcn_wmma_f32_16x16x32_bf16(false, a, false, b, (short)0, c, false, false);
    }
    const int col = n0 + (lane & 15);
    const float bv = bias[col];
    #pragma unroll
    for (int r = 0; r < 8; ++r) {
        int row = m0 + r + (hi ? 8 : 0);
        C[(size_t)row * ldc + col] = c[r] + bv;
    }
}

// Persistent recurrence kernel: NB blocks x 256 threads. Block b owns output
// rows [b*64, b*64+64). 8 waves: wave w -> row tile (w&3), K-half (w>>2).
// A operand is h_{t-1} broadcast to all 16 M-rows, so every D row equals y.
__global__ __launch_bounds__(256)
void rnn_recurrence(const float* __restrict__ xh,
                    const __bf16* __restrict__ w2,
                    const float* __restrict__ bh,
                    __bf16* __restrict__ h_all,
                    int* __restrict__ bar) {
    __shared__ __align__(16) __bf16 hs[HH];   // 8 KB: h_{t-1} staged in LDS
    __shared__ float ybuf[64];                 // cross-wave K reduction

    const int wave  = threadIdx.x >> 5;
    const int lane  = threadIdx.x & 31;
    const int tile  = wave & 3;
    const int khalf = wave >> 2;
    const int n0    = blockIdx.x * 64 + tile * 16;
    const bool hi   = lane >= 16;
    const size_t brow = (size_t)(n0 + (lane & 15)) * HH + (hi ? 16 : 0);
    const int kbeg  = khalf * (HH / 2);
    const int kend  = kbeg + (HH / 2);
    const int aoff  = hi ? 16 : 0;

    for (int t = 0; t < TT; ++t) {
        // stage h_{t-1} (row t of h_all) into LDS: 512 uint4, 2 per thread
        {
            const uint4* src = (const uint4*)(h_all + (size_t)t * HH);
            uint4* dst = (uint4*)hs;
            dst[threadIdx.x]       = src[threadIdx.x];
            dst[threadIdx.x + 256] = src[threadIdx.x + 256];
        }
        __syncthreads();

        v8f c = {};
        #pragma unroll 4
        for (int k0 = kbeg; k0 < kend; k0 += 32) {
            v8bf alo = *(const v8bf*)(hs + k0 + aoff);
            v8bf ahi = *(const v8bf*)(hs + k0 + aoff + 8);
            v16bf a = __builtin_shufflevector(alo, ahi, 0,1,2,3,4,5,6,7,8,9,10,11,12,13,14,15);
            v16bf b = *(const v16bf*)(w2 + brow + k0);
            c = __builtin_amdgcn_wmma_f32_16x16x32_bf16(false, a, false, b, (short)0, c, false, false);
        }

        // reduce the two K-halves; lanes 0..15 of c[0] hold y[n0..n0+15]
        if (khalf == 1 && !hi) ybuf[tile * 16 + lane] = c[0];
        __syncthreads();
        if (khalf == 0 && !hi) {
            int n = n0 + lane;
            float y = c[0] + ybuf[tile * 16 + lane]
                    + xh[(size_t)t * HH + n] + bh[n];
            h_all[(size_t)(t + 1) * HH + n] = (__bf16)tanhf(y);
        }

        // device-wide barrier (monotonic counter; fresh h row address each
        // step, so no stale-line hazard on the subsequent reads)
        __threadfence();
        __syncthreads();
        if (threadIdx.x == 0) {
            atomicAdd(bar, 1);
            const int target = NB * (t + 1);
            while (__hip_atomic_load(bar, __ATOMIC_ACQUIRE, __HIP_MEMORY_SCOPE_AGENT) < target)
                __builtin_amdgcn_s_sleep(1);
        }
        __syncthreads();
    }
}

extern "C" void kernel_launch(void* const* d_in, const int* in_sizes, int n_in,
                              void* d_out, int out_size, void* d_ws, size_t ws_size,
                              hipStream_t stream) {
    const float* x      = (const float*)d_in[0];   // [T,L]
    const float* W_i2h  = (const float*)d_in[1];   // [H,L]
    const float* b_i2h  = (const float*)d_in[2];   // [H]
    const float* W_h2h  = (const float*)d_in[3];   // [H,H]
    const float* b_h2h  = (const float*)d_in[4];   // [H]
    const float* W_h2o  = (const float*)d_in[5];   // [L,H]
    const float* b_h2o  = (const float*)d_in[6];   // [L]
    float* out = (float*)d_out;                    // [T,L]

    char* ws = (char*)d_ws;
    float*  xh   = (float*)(ws + OFF_XH);
    __bf16* hall = (__bf16*)(ws + OFF_HALL);
    __bf16* w2   = (__bf16*)(ws + OFF_W2);
    __bf16* xbf  = (__bf16*)(ws + OFF_XBF);
    __bf16* w1   = (__bf16*)(ws + OFF_W1);
    __bf16* w3   = (__bf16*)(ws + OFF_W3);
    int*    bar  = (int*)(ws + OFF_BAR);

    // barrier counter = 0, h_{-1} = 0 (row 0 of h_all)
    hipMemsetAsync(ws + OFF_BAR, 0, 256, stream);
    hipMemsetAsync(ws + OFF_HALL, 0, (size_t)HH * 2, stream);

    // fp32 -> bf16 weight/input conversions
    cvt_f32_bf16<<<(HH * HH + 255) / 256, 256, 0, stream>>>(W_h2h, w2, HH * HH);
    cvt_f32_bf16<<<(TT * LL + 255) / 256, 256, 0, stream>>>(x, xbf, TT * LL);
    cvt_f32_bf16<<<(HH * LL + 255) / 256, 256, 0, stream>>>(W_i2h, w1, HH * LL);
    cvt_f32_bf16<<<(LL * HH + 255) / 256, 256, 0, stream>>>(W_h2o, w3, LL * HH);

    // Phase 1: xh[T,H] = x[T,L] @ W_i2h^T + b_i2h
    //   tiles: 128 x 256 = 32768, 8 waves/block -> 4096 blocks
    wmma_gemm_bias<<<4096, 256, 0, stream>>>(xbf, LL, w1, LL, b_i2h, xh, HH,
                                             TT / 16, LL);

    // Phase 2: sequential scan, h_all rows 1..T
    rnn_recurrence<<<NB, 256, 0, stream>>>(xh, w2, b_h2h, hall, bar);

    // Phase 3: out[T,L] = h_all[1..T][H] @ W_h2o^T + b_h2o
    //   tiles: 128 x 32 = 4096, 8 waves/block -> 512 blocks
    wmma_gemm_bias<<<512, 256, 0, stream>>>(hall + HH, HH, w3, HH, b_h2o, out, LL,
                                            TT / 16, HH);

    (void)in_sizes; (void)n_in; (void)out_size; (void)ws_size;
}
